// GaussianActionDistGenerator_60387240182361
// MI455X (gfx1250) — matrix operand order repaired and checked by reading, not compiled, not run on hardware
//
#include <hip/hip_runtime.h>

#define LSEQ 2048
#define DDIM 128
#define NB   8
#define NROW (NB * LSEQ)   // 16384 rows per tensor

typedef __attribute__((ext_vector_type(16))) __bf16 v16bf;
typedef __attribute__((ext_vector_type(8)))  float  v8f;

// ---------------------------------------------------------------------------
// Hardware tanh when available (V_TANH_F32 is a CDNA5 TRANS32 op), else ocml.
// ---------------------------------------------------------------------------
__device__ __forceinline__ float fast_tanh(float x)
{
#if __has_builtin(__builtin_amdgcn_tanhf)
    return __builtin_amdgcn_tanhf(x);
#elif __has_builtin(__builtin_amdgcn_tanh_f32)
    return __builtin_amdgcn_tanh_f32(x);
#else
    return tanhf(x);
#endif
}

// ---------------------------------------------------------------------------
// Stage 1: linear projections in f32, results stored as bf16 for WMMA stage.
// One block (128 threads) per row; thread t computes output feature t for the
// mean and std projections of that row. Weights are 64 KB each -> L2 resident.
// ---------------------------------------------------------------------------
__global__ __launch_bounds__(128)
void proj_kernel(const float* __restrict__ query,
                 const float* __restrict__ key,
                 const float* __restrict__ Wq_mean, const float* __restrict__ bq_mean,
                 const float* __restrict__ Wk_mean, const float* __restrict__ bk_mean,
                 const float* __restrict__ Wq_std,  const float* __restrict__ bq_std,
                 const float* __restrict__ Wk_std,  const float* __restrict__ bk_std,
                 __bf16* __restrict__ qm, __bf16* __restrict__ qs,
                 __bf16* __restrict__ km, __bf16* __restrict__ ks)
{
    __shared__ float xrow[DDIM];
    const int t     = threadIdx.x;
    const int rowId = blockIdx.x;            // 0 .. 2*NROW-1
    const bool isKey = rowId >= NROW;
    const int r = isKey ? (rowId - NROW) : rowId;

    const float* __restrict__ src = isKey ? key : query;
    xrow[t] = src[(size_t)r * DDIM + t];
    __syncthreads();

    const float* __restrict__ Wm = isKey ? Wk_mean : Wq_mean;
    const float* __restrict__ Ws = isKey ? Wk_std  : Wq_std;
    float am = (isKey ? bk_mean : bq_mean)[t];
    float as = (isKey ? bk_std  : bq_std )[t];

#pragma unroll 8
    for (int i = 0; i < DDIM; ++i) {
        const float x = xrow[i];
        am = fmaf(x, Wm[i * DDIM + t], am);   // column t access: coalesced
        as = fmaf(x, Ws[i * DDIM + t], as);
    }

    __bf16* __restrict__ om = isKey ? km : qm;
    __bf16* __restrict__ os = isKey ? ks : qs;
    om[(size_t)r * DDIM + t] = (__bf16)am;
    os[(size_t)r * DDIM + t] = (__bf16)as;
}

// ---------------------------------------------------------------------------
// Stage 2: S = (Q @ K^T) / sqrt(D) via v_wmma_f32_16x16x32_bf16, fused
// tanh clipping + masking, NT stores for the 268 MB of output.
//
// A-fragment (16x32 bf16, ISA 7.12.2): lane l holds row (l&15);
//   VGPR0-3 = K[8h .. 8h+7], VGPR4-7 = K[16+8h .. 16+8h+7], h = l>>4.
// B-fragment (32x16 bf16): lane l holds column (l&15);
//   VGPR0-3 = K[8h .. 8h+7], VGPR4-7 = K[16+8h .. 16+8h+7] (same as A).
// ---------------------------------------------------------------------------
__device__ __forceinline__ v16bf load_frag_a(const __bf16* __restrict__ base,
                                             int row0, int kk, int lane)
{
    const int r = lane & 15, h = lane >> 4;
    const __bf16* p = base + (size_t)(row0 + r) * DDIM + kk + 8 * h;
    union { v16bf v; uint4 u[2]; } f;
    f.u[0] = *(const uint4*)(p);        // K = kk+8h   .. +7
    f.u[1] = *(const uint4*)(p + 16);   // K = kk+16+8h .. +7
    return f.v;
}

__device__ __forceinline__ v16bf load_frag_b(const __bf16* __restrict__ base,
                                             int col0, int kk, int lane)
{
    const int c = lane & 15, h = lane >> 4;
    const __bf16* p = base + (size_t)(col0 + c) * DDIM + kk + 8 * h;
    union { v16bf v; uint4 u[2]; } f;
    f.u[0] = *(const uint4*)(p);        // K = kk+8h   .. +7
    f.u[1] = *(const uint4*)(p + 16);   // K = kk+16+8h .. +7
    return f.v;
}

#define WMMA_BF16(A, B, C) \
    __builtin_amdgcn_wmma_f32_16x16x32_bf16(false, (A), false, (B), (short)0, (C), false, false)

__global__ __launch_bounds__(256)
void score_kernel(const __bf16* __restrict__ qm, const __bf16* __restrict__ qs,
                  const __bf16* __restrict__ km, const __bf16* __restrict__ ks,
                  const int*    __restrict__ mask,
                  float* __restrict__ out_mean, float* __restrict__ out_std)
{
    const int lane = threadIdx.x & 31;
    const int wave = threadIdx.x >> 5;      // 8 waves
    const int wm   = wave >> 1;             // 0..3
    const int wn   = wave & 1;              // 0..1
    const int b    = blockIdx.z;
    const int m_base = blockIdx.y * 128 + wm * 32;
    const int n_base = blockIdx.x * 64  + wn * 32;

    const size_t bo = (size_t)b * LSEQ * DDIM;
    const __bf16* __restrict__ qmb = qm + bo;
    const __bf16* __restrict__ qsb = qs + bo;
    const __bf16* __restrict__ kmb = km + bo;
    const __bf16* __restrict__ ksb = ks + bo;

    const v8f zero = {0.f, 0.f, 0.f, 0.f, 0.f, 0.f, 0.f, 0.f};
    v8f accm[2][2] = {{zero, zero}, {zero, zero}};
    v8f accs[2][2] = {{zero, zero}, {zero, zero}};

    // ---- mean GEMM: 32x32 wave tile, K = 128 in 4 steps of 32 ----
#pragma unroll
    for (int kk = 0; kk < DDIM; kk += 32) {
        const v16bf a0 = load_frag_a(qmb, m_base,      kk, lane);
        const v16bf a1 = load_frag_a(qmb, m_base + 16, kk, lane);
        const v16bf b0 = load_frag_b(kmb, n_base,      kk, lane);
        const v16bf b1 = load_frag_b(kmb, n_base + 16, kk, lane);
        accm[0][0] = WMMA_BF16(a0, b0, accm[0][0]);
        accm[0][1] = WMMA_BF16(a0, b1, accm[0][1]);
        accm[1][0] = WMMA_BF16(a1, b0, accm[1][0]);
        accm[1][1] = WMMA_BF16(a1, b1, accm[1][1]);
    }

    // ---- std GEMM ----
#pragma unroll
    for (int kk = 0; kk < DDIM; kk += 32) {
        const v16bf a0 = load_frag_a(qsb, m_base,      kk, lane);
        const v16bf a1 = load_frag_a(qsb, m_base + 16, kk, lane);
        const v16bf b0 = load_frag_b(ksb, n_base,      kk, lane);
        const v16bf b1 = load_frag_b(ksb, n_base + 16, kk, lane);
        accs[0][0] = WMMA_BF16(a0, b0, accs[0][0]);
        accs[0][1] = WMMA_BF16(a0, b1, accs[0][1]);
        accs[1][0] = WMMA_BF16(a1, b0, accs[1][0]);
        accs[1][1] = WMMA_BF16(a1, b1, accs[1][1]);
    }

    // ---- fused epilogue: one mask read feeds both outputs ----
    const float scale = 0.08838834764831845f;   // 1/sqrt(128)
    const int r = lane & 15, h = lane >> 4;
    const size_t bout = (size_t)b * LSEQ * LSEQ;

#pragma unroll
    for (int ti = 0; ti < 2; ++ti) {
#pragma unroll
        for (int tj = 0; tj < 2; ++tj) {
            const int col = n_base + tj * 16 + r;
#pragma unroll
            for (int i = 0; i < 8; ++i) {
                const int row = m_base + ti * 16 + h * 8 + i;   // C layout: M = 8h+i
                const size_t idx = bout + (size_t)row * LSEQ + col;
                const int mv = __builtin_nontemporal_load(&mask[idx]);
                const float tm = fast_tanh(accm[ti][tj][i] * scale);
                const float ts = fast_tanh(accs[ti][tj][i] * scale);
                const float vm = mv ? 0.5f * (tm + 1.0f) : 0.0f;
                // (CLIP_STD - U) * (t-1)/2 - U  ==  4*(t-1) - 2
                const float vs = mv ? (4.0f * (ts - 1.0f) - 2.0f) : -10.0f;
                __builtin_nontemporal_store(vm, &out_mean[idx]);
                __builtin_nontemporal_store(vs, &out_std[idx]);
            }
        }
    }
}

// ---------------------------------------------------------------------------
extern "C" void kernel_launch(void* const* d_in, const int* in_sizes, int n_in,
                              void* d_out, int out_size, void* d_ws, size_t ws_size,
                              hipStream_t stream)
{
    const float* query   = (const float*)d_in[0];
    const float* key     = (const float*)d_in[1];
    const int*   mask    = (const int*)  d_in[2];
    const float* Wq_mean = (const float*)d_in[3];
    const float* bq_mean = (const float*)d_in[4];
    const float* Wk_mean = (const float*)d_in[5];
    const float* bk_mean = (const float*)d_in[6];
    const float* Wq_std  = (const float*)d_in[7];
    const float* bq_std  = (const float*)d_in[8];
    const float* Wk_std  = (const float*)d_in[9];
    const float* bk_std  = (const float*)d_in[10];

    const size_t NTOT = (size_t)NROW * DDIM;      // 2M elems per projection
    __bf16* qm = (__bf16*)d_ws;                   // 4 * 4 MB bf16 scratch
    __bf16* qs = qm + NTOT;
    __bf16* km = qs + NTOT;
    __bf16* ks = km + NTOT;

    float* out_mean = (float*)d_out;
    float* out_std  = out_mean + (size_t)NB * LSEQ * LSEQ;

    proj_kernel<<<2 * NROW, 128, 0, stream>>>(
        query, key, Wq_mean, bq_mean, Wk_mean, bk_mean,
        Wq_std, bq_std, Wk_std, bk_std, qm, qs, km, ks);

    dim3 grid(LSEQ / 64, LSEQ / 128, NB);         // (32, 16, 8)
    score_kernel<<<grid, 256, 0, stream>>>(qm, qs, km, ks, mask, out_mean, out_std);
}
